// RelationalTransformerUpdate_7928509628690
// MI455X (gfx1250) — matrix operand order repaired and checked by reading, not compiled, not run on hardware
//
#include <hip/hip_runtime.h>

// Problem constants (fixed by the reference)
constexpr int Bc  = 2;
constexpr int Sc  = 512;
constexpr int Dc  = 512;
constexpr int Hc  = 8;
constexpr int Lc  = 4;
constexpr int FFc = 2048;
constexpr int Rc  = 37;
constexpr int DKc = 64;

typedef __attribute__((ext_vector_type(16))) __bf16 v16bf;
typedef __attribute__((ext_vector_type(8)))  float  v8f;

__device__ __forceinline__ unsigned short f2bf(float f) {
    unsigned u = __float_as_uint(f);
    unsigned r = u + 0x7FFFu + ((u >> 16) & 1u);   // round-to-nearest-even
    return (unsigned short)(r >> 16);
}
__device__ __forceinline__ float bf2f(unsigned short h) {
    return __uint_as_float(((unsigned)h) << 16);
}

union BfFrag { uint4 u[2]; v16bf v; };

// Async global->LDS staging (CDNA5 VGLOBAL async ops, tracked by ASYNCcnt).
// Generic pointers to __shared__ carry the wave-relative LDS offset in their
// low 32 bits (ISA 10.2), which is what VDST expects.
__device__ __forceinline__ void async_load_b128(const void* gaddr, void* lds) {
    unsigned l = (unsigned)(unsigned long long)lds;
    unsigned long long g = (unsigned long long)gaddr;
    asm volatile("global_load_async_to_lds_b128 %0, %1, off"
                 :: "v"(l), "v"(g) : "memory");
}
// Async loads complete in order, so waiting ASYNCcnt<=8 with 16 outstanding
// guarantees the older 8 (the current buffer's chunk) have landed.
__device__ __forceinline__ void wait_async_le0() {
    asm volatile("s_wait_asynccnt 0x0" ::: "memory");
}
__device__ __forceinline__ void wait_async_le8() {
    asm volatile("s_wait_asynccnt 0x8" ::: "memory");
}

__device__ __forceinline__ v8f wmma_bf16(const v16bf& a, const v16bf& b, v8f c) {
    return __builtin_amdgcn_wmma_f32_16x16x32_bf16(false, a, false, b,
                                                   (short)0, c, false, false);
}

// ---------------------------------------------------------------------------
// Weight transpose + convert: dst[n*K + k] = bf16(src[k*N + n])   (B^T layout)
// ---------------------------------------------------------------------------
__global__ void tconv_kernel(unsigned short* __restrict__ dst,
                             const float* __restrict__ src, int K, int N) {
    int i = blockIdx.x * blockDim.x + threadIdx.x;
    if (i >= K * N) return;
    int n = i / K, k = i % K;
    dst[i] = f2bf(src[(size_t)k * N + n]);
}

__global__ void copy_f32_kernel(float* __restrict__ dst,
                                const float* __restrict__ src, int n) {
    int i = blockIdx.x * blockDim.x + threadIdx.x;
    if (i < n) dst[i] = src[i];
}

// ---------------------------------------------------------------------------
// RAT-SQL LayerNorm: g*(x-mu)/(std_unbiased + eps) + b, over D=512 per row.
// ---------------------------------------------------------------------------
__global__ void layernorm_kernel(const float* __restrict__ x,
                                 const float* __restrict__ g,
                                 const float* __restrict__ b,
                                 unsigned short* __restrict__ outH,
                                 float* __restrict__ outF) {
    __shared__ float red[256];
    int row = blockIdx.x, t = threadIdx.x;
    const float* xr = x + (size_t)row * Dc;
    float v0 = xr[t], v1 = xr[t + 256];

    red[t] = v0 + v1; __syncthreads();
    for (int s = 128; s > 0; s >>= 1) { if (t < s) red[t] += red[t + s]; __syncthreads(); }
    float mu = red[0] * (1.0f / Dc); __syncthreads();

    float d0 = v0 - mu, d1 = v1 - mu;
    red[t] = d0 * d0 + d1 * d1; __syncthreads();
    for (int s = 128; s > 0; s >>= 1) { if (t < s) red[t] += red[t + s]; __syncthreads(); }
    float var = red[0] * (1.0f / (Dc - 1));          // unbiased (ddof=1)
    float inv = 1.0f / (sqrtf(var) + 1e-6f);

    float o0 = g[t]       * d0 * inv + b[t];
    float o1 = g[t + 256] * d1 * inv + b[t + 256];
    if (outH) { outH[(size_t)row * Dc + t] = f2bf(o0); outH[(size_t)row * Dc + t + 256] = f2bf(o1); }
    if (outF) { outF[(size_t)row * Dc + t] = o0;       outF[(size_t)row * Dc + t + 256] = o1; }
}

// ---------------------------------------------------------------------------
// Flat WMMA GEMM with double-buffered async-LDS staging.
// Block = 128 threads (4 waves) computing a 64x64 macro-tile; wave w owns
// rows [w*16, w*16+16) x all 64 cols (4 f32 accumulators).
// K is consumed in 64-wide chunks: chunk k+1 is staged with
// global_load_async_to_lds_b128 (ASYNCcnt) while chunk k is consumed via
// ds_load_b128; s_wait_asynccnt 8 exploits in-order async completion.
// Requires M%64==0, N%64==0, K%64==0 (true for all uses here).
// ---------------------------------------------------------------------------
__global__ __launch_bounds__(128)
void gemm_wmma_kernel(const unsigned short* __restrict__ A, int lda,
                      const unsigned short* __restrict__ BT, int ldb,
                      const float* __restrict__ bias,
                      float* __restrict__ outF, int ldo, int addto,
                      unsigned short* __restrict__ outH, int ldoh,
                      int M, int N, int K, int relu) {
    constexpr int KC = 64;
    __shared__ __align__(16) unsigned short shA[2][64 * KC];
    __shared__ __align__(16) unsigned short shB[2][64 * KC];

    int lane = threadIdx.x;            // 0..31
    int wv   = threadIdx.y;            // 0..3
    int tid  = wv * 32 + lane;         // 0..127
    int g    = lane >> 4;
    int ml   = lane & 15;
    int m0   = blockIdx.x * 64;
    int n0   = blockIdx.y * 64;

    // stage one 64xKC A tile + one 64xKC B tile: 512 16B chunks each,
    // 128 threads x 4 iterations x 2 async loads = 8 async ops per wave.
    auto stage = [&](int buf, int k0) {
#pragma unroll
        for (int ch = tid; ch < 512; ch += 128) {
            int r  = ch >> 3;            // 0..63
            int cc = (ch & 7) * 8;       // half offset within row
            async_load_b128(A  + (size_t)(m0 + r) * lda + k0 + cc, &shA[buf][r * KC + cc]);
            async_load_b128(BT + (size_t)(n0 + r) * ldb + k0 + cc, &shB[buf][r * KC + cc]);
        }
    };

    v8f c0 = {}, c1 = {}, c2 = {}, c3 = {};

    stage(0, 0);
    int cur = 0;
    for (int k0 = 0; k0 < K; k0 += KC) {
        bool has_next = (k0 + KC) < K;
        if (has_next) stage(cur ^ 1, k0 + KC);
        if (has_next) wait_async_le8(); else wait_async_le0();
        __syncthreads();

        const unsigned short* sA = shA[cur];
        const unsigned short* sB = shB[cur];
#pragma unroll
        for (int kk = 0; kk < KC; kk += 32) {
            BfFrag a;
            const unsigned short* ar = sA + (wv * 16 + ml) * KC + kk;
            a.u[0] = *(const uint4*)(ar + 16 * g);
            a.u[1] = *(const uint4*)(ar + 16 * g + 8);
            BfFrag b0, b1, b2, b3;
            const unsigned short* br0 = sB + (0 * 16 + ml) * KC + kk + 16 * g;
            const unsigned short* br1 = sB + (1 * 16 + ml) * KC + kk + 16 * g;
            const unsigned short* br2 = sB + (2 * 16 + ml) * KC + kk + 16 * g;
            const unsigned short* br3 = sB + (3 * 16 + ml) * KC + kk + 16 * g;
            b0.u[0] = *(const uint4*)(br0); b0.u[1] = *(const uint4*)(br0 + 8);
            b1.u[0] = *(const uint4*)(br1); b1.u[1] = *(const uint4*)(br1 + 8);
            b2.u[0] = *(const uint4*)(br2); b2.u[1] = *(const uint4*)(br2 + 8);
            b3.u[0] = *(const uint4*)(br3); b3.u[1] = *(const uint4*)(br3 + 8);
            c0 = wmma_bf16(a.v, b0.v, c0);
            c1 = wmma_bf16(a.v, b1.v, c1);
            c2 = wmma_bf16(a.v, b2.v, c2);
            c3 = wmma_bf16(a.v, b3.v, c3);
        }
        __syncthreads();
        cur ^= 1;
    }

    v8f acc[4] = {c0, c1, c2, c3};
#pragma unroll
    for (int j = 0; j < 4; ++j) {
        int ncol = n0 + j * 16 + ml;
        float bv = bias ? bias[ncol] : 0.0f;
#pragma unroll
        for (int e = 0; e < 8; ++e) {
            int mr = m0 + wv * 16 + e + 8 * g;
            float v = acc[j][e] + bv;
            if (relu) v = fmaxf(v, 0.0f);
            if (outF) {
                size_t idx = (size_t)mr * ldo + ncol;
                outF[idx] = addto ? (outF[idx] + v) : v;
            }
            if (outH) outH[(size_t)mr * ldoh + ncol] = f2bf(v);
        }
    }
    (void)M; (void)N;
}

// ---------------------------------------------------------------------------
// Per-head V transpose: vt[b,h,d,s] = v[b*S+s, h*DK+d]   (bf16 -> bf16)
// ---------------------------------------------------------------------------
__global__ void vt_kernel(unsigned short* __restrict__ vt,
                          const unsigned short* __restrict__ vh) {
    int i = blockIdx.x * blockDim.x + threadIdx.x;
    if (i >= Bc * Sc * Dc) return;
    int col = i % Dc, rowg = i / Dc;       // rowg = b*S + s
    int b = rowg / Sc, s = rowg % Sc;
    int h = col / DKc, d = col % DKc;
    vt[(((size_t)(b * Hc + h) * DKc + d) * Sc) + s] = vh[i];
}

// ---------------------------------------------------------------------------
// p[b,h,s,r] = q[b,h,s,:] . rel_k_emb[r,:]   (R=37, DK=64)
// ---------------------------------------------------------------------------
__global__ void pq_kernel(const unsigned short* __restrict__ qh,
                          const float* __restrict__ relk,
                          float* __restrict__ p) {
    __shared__ float qrow[DKc];
    int row = blockIdx.x;                  // row = (b*H+h)*S + s
    int z = row / Sc, s = row % Sc;
    int b = z / Hc,  h = z % Hc;
    int t = threadIdx.x;                   // 0..63
    qrow[t] = bf2f(qh[((size_t)(b * Sc + s) * Dc) + h * DKc + t]);
    __syncthreads();
    if (t < Rc) {
        const float* e = relk + (size_t)t * DKc;
        float acc = 0.0f;
#pragma unroll
        for (int d = 0; d < DKc; ++d) acc += qrow[d] * e[d];
        p[(size_t)row * Rc + t] = acc;
    }
}

// ---------------------------------------------------------------------------
// scores[b,h,q,k] = (q.kT + p[b,h,q,rel[b,q,k]]) * scale, mask -> -1e9
// grid = (S/16, S/256, B*H), block = (32,4): each wave owns 16x64 of scores.
// ---------------------------------------------------------------------------
__global__ void scores_wmma_kernel(const unsigned short* __restrict__ qh,
                                   const unsigned short* __restrict__ kh,
                                   const float* __restrict__ p,
                                   const int* __restrict__ relations,
                                   const int* __restrict__ mask,
                                   float* __restrict__ scores) {
    int lane = threadIdx.x;
    int z = blockIdx.z;                    // b*H + h
    int b = z / Hc, h = z % Hc;
    int mt  = blockIdx.x;
    int n64 = blockIdx.y * 4 + threadIdx.y;   // 64-wide N slab
    int g  = lane >> 4, ml = lane & 15;

    const unsigned short* arow = qh + (size_t)(b * Sc + mt * 16 + ml) * Dc + h * DKc;
    const unsigned short* br[4];
#pragma unroll
    for (int j = 0; j < 4; ++j)
        br[j] = kh + (size_t)(b * Sc + n64 * 64 + j * 16 + ml) * Dc + h * DKc;

    v8f c0 = {}, c1 = {}, c2 = {}, c3 = {};
#pragma unroll
    for (int k0 = 0; k0 < DKc; k0 += 32) {
        BfFrag a, b0, b1, b2, b3;
        a.u[0] = *(const uint4*)(arow + k0 + 16 * g);
        a.u[1] = *(const uint4*)(arow + k0 + 16 * g + 8);
        b0.u[0] = *(const uint4*)(br[0] + k0 + 16 * g); b0.u[1] = *(const uint4*)(br[0] + k0 + 16 * g + 8);
        b1.u[0] = *(const uint4*)(br[1] + k0 + 16 * g); b1.u[1] = *(const uint4*)(br[1] + k0 + 16 * g + 8);
        b2.u[0] = *(const uint4*)(br[2] + k0 + 16 * g); b2.u[1] = *(const uint4*)(br[2] + k0 + 16 * g + 8);
        b3.u[0] = *(const uint4*)(br[3] + k0 + 16 * g); b3.u[1] = *(const uint4*)(br[3] + k0 + 16 * g + 8);
        c0 = wmma_bf16(a.v, b0.v, c0);
        c1 = wmma_bf16(a.v, b1.v, c1);
        c2 = wmma_bf16(a.v, b2.v, c2);
        c3 = wmma_bf16(a.v, b3.v, c3);
    }

    const float scale = 0.125f;            // 1/sqrt(64)
    v8f acc[4] = {c0, c1, c2, c3};
#pragma unroll
    for (int j = 0; j < 4; ++j) {
        int ncol = n64 * 64 + j * 16 + ml;
#pragma unroll
        for (int e = 0; e < 8; ++e) {
            int mr = mt * 16 + e + 8 * g;
            size_t pair = (size_t)(b * Sc + mr) * Sc + ncol;
            int rel = relations[pair];
            float v = (acc[j][e] + p[((size_t)z * Sc + mr) * Rc + rel]) * scale;
            if (mask[pair] == 0) v = -1e9f;
            scores[((size_t)z * Sc + mr) * Sc + ncol] = v;
        }
    }
}

// ---------------------------------------------------------------------------
// Row softmax over S=512; writes normalized f32 back in place + bf16 copy.
// ---------------------------------------------------------------------------
__global__ void softmax_kernel(float* __restrict__ scores,
                               unsigned short* __restrict__ attnh) {
    __shared__ float red[256];
    int row = blockIdx.x, t = threadIdx.x;
    float* sr = scores + (size_t)row * Sc;
    float a0 = sr[t], a1 = sr[t + 256];

    red[t] = fmaxf(a0, a1); __syncthreads();
    for (int s = 128; s > 0; s >>= 1) { if (t < s) red[t] = fmaxf(red[t], red[t + s]); __syncthreads(); }
    float mx = red[0]; __syncthreads();

    float e0 = expf(a0 - mx), e1 = expf(a1 - mx);
    red[t] = e0 + e1; __syncthreads();
    for (int s = 128; s > 0; s >>= 1) { if (t < s) red[t] += red[t + s]; __syncthreads(); }
    float inv = 1.0f / red[0];

    float p0 = e0 * inv, p1 = e1 * inv;
    sr[t] = p0; sr[t + 256] = p1;
    attnh[(size_t)row * Sc + t]       = f2bf(p0);
    attnh[(size_t)row * Sc + t + 256] = f2bf(p1);
}

// ---------------------------------------------------------------------------
// Relation buckets: w[b,h,s,r] = sum_k attn[b,h,s,k] * (rel[b,s,k]==r)
// ---------------------------------------------------------------------------
__global__ void bucket_kernel(const float* __restrict__ attn,
                              const int* __restrict__ relations,
                              float* __restrict__ w) {
    __shared__ float wl[Rc];
    int row = blockIdx.x;                  // (b*H+h)*S + s
    int z = row / Sc, s = row % Sc;
    int b = z / Hc;
    int t = threadIdx.x;
    if (t < Rc) wl[t] = 0.0f;
    __syncthreads();
    const float* ar = attn + (size_t)row * Sc;
    const int* rr = relations + (size_t)(b * Sc + s) * Sc;
    for (int k = t; k < Sc; k += blockDim.x) atomicAdd(&wl[rr[k]], ar[k]);
    __syncthreads();
    if (t < Rc) w[(size_t)row * Rc + t] = wl[t];
}

// ---------------------------------------------------------------------------
// o[b,h,s,d] = attn @ v   (per head; A=attn bf16 [S,S], BT=vt bf16 [DK,S])
// grid = (S/64, 1, B*H), block = (32,4): each wave owns 16 rows x DK=64 cols.
// ---------------------------------------------------------------------------
__global__ void av_wmma_kernel(const unsigned short* __restrict__ attnh,
                               const unsigned short* __restrict__ vt,
                               float* __restrict__ o) {
    int lane = threadIdx.x;
    int z  = blockIdx.z;
    int mt = blockIdx.x * 4 + threadIdx.y;
    int g  = lane >> 4, ml = lane & 15;

    const unsigned short* arow = attnh + (size_t)z * Sc * Sc + (size_t)(mt * 16 + ml) * Sc;
    const unsigned short* br[4];
#pragma unroll
    for (int j = 0; j < 4; ++j)
        br[j] = vt + (size_t)z * DKc * Sc + (size_t)(j * 16 + ml) * Sc;

    v8f c0 = {}, c1 = {}, c2 = {}, c3 = {};
    for (int k0 = 0; k0 < Sc; k0 += 32) {
        BfFrag a, b0, b1, b2, b3;
        a.u[0] = *(const uint4*)(arow + k0 + 16 * g);
        a.u[1] = *(const uint4*)(arow + k0 + 16 * g + 8);
        b0.u[0] = *(const uint4*)(br[0] + k0 + 16 * g); b0.u[1] = *(const uint4*)(br[0] + k0 + 16 * g + 8);
        b1.u[0] = *(const uint4*)(br[1] + k0 + 16 * g); b1.u[1] = *(const uint4*)(br[1] + k0 + 16 * g + 8);
        b2.u[0] = *(const uint4*)(br[2] + k0 + 16 * g); b2.u[1] = *(const uint4*)(br[2] + k0 + 16 * g + 8);
        b3.u[0] = *(const uint4*)(br[3] + k0 + 16 * g); b3.u[1] = *(const uint4*)(br[3] + k0 + 16 * g + 8);
        if (k0 + 32 < Sc) {
            __builtin_prefetch(arow + k0 + 96, 0, 3);
            __builtin_prefetch(br[0] + k0 + 96, 0, 3);
        }
        c0 = wmma_bf16(a.v, b0.v, c0);
        c1 = wmma_bf16(a.v, b1.v, c1);
        c2 = wmma_bf16(a.v, b2.v, c2);
        c3 = wmma_bf16(a.v, b3.v, c3);
    }

    v8f acc[4] = {c0, c1, c2, c3};
#pragma unroll
    for (int j = 0; j < 4; ++j) {
        int ncol = j * 16 + ml;
#pragma unroll
        for (int e = 0; e < 8; ++e) {
            int mr = mt * 16 + e + 8 * g;
            o[((size_t)z * Sc + mr) * DKc + ncol] = acc[j][e];
        }
    }
}

// ---------------------------------------------------------------------------
// o += w @ rel_v_emb; scatter heads back to [B*S, D] bf16 for the O-proj GEMM
// ---------------------------------------------------------------------------
__global__ void orv_kernel(const float* __restrict__ o,
                           const float* __restrict__ w,
                           const float* __restrict__ relv,
                           unsigned short* __restrict__ oh) {
    int i = blockIdx.x * blockDim.x + threadIdx.x;
    if (i >= Bc * Hc * Sc * DKc) return;
    int d = i & (DKc - 1);
    int row = i >> 6;                      // (b*H+h)*S + s
    int z = row / Sc, s = row % Sc;
    int b = z / Hc,  h = z % Hc;
    float acc = o[i];
    const float* wr = w + (size_t)row * Rc;
#pragma unroll
    for (int r = 0; r < Rc; ++r) acc += wr[r] * relv[(size_t)r * DKc + d];
    oh[((size_t)(b * Sc + s) * Dc) + h * DKc + d] = f2bf(acc);
}

// ---------------------------------------------------------------------------
// Host orchestration
// ---------------------------------------------------------------------------
extern "C" void kernel_launch(void* const* d_in, const int* in_sizes, int n_in,
                              void* d_out, int out_size, void* d_ws, size_t ws_size,
                              hipStream_t stream) {
    const float* enc       = (const float*)d_in[0];
    const int*   relations = (const int*)  d_in[1];
    const int*   mask      = (const int*)  d_in[2];
    const float* Wq        = (const float*)d_in[3];
    const float* bq        = (const float*)d_in[4];
    const float* Wk        = (const float*)d_in[5];
    const float* bk        = (const float*)d_in[6];
    const float* Wv        = (const float*)d_in[7];
    const float* bv        = (const float*)d_in[8];
    const float* Wo        = (const float*)d_in[9];
    const float* bo        = (const float*)d_in[10];
    const float* relk      = (const float*)d_in[11];
    const float* relv      = (const float*)d_in[12];
    const float* ln1_g     = (const float*)d_in[13];
    const float* ln1_b     = (const float*)d_in[14];
    const float* W1        = (const float*)d_in[15];
    const float* b1        = (const float*)d_in[16];
    const float* W2        = (const float*)d_in[17];
    const float* b2        = (const float*)d_in[18];
    const float* ln2_g     = (const float*)d_in[19];
    const float* ln2_b     = (const float*)d_in[20];
    const float* lnf_g     = (const float*)d_in[21];
    const float* lnf_b     = (const float*)d_in[22];

    // workspace carve (256B aligned)
    char* base = (char*)d_ws;
    size_t off = 0;
    auto carve = [&](size_t bytes) -> char* {
        char* p = base + off;
        off += (bytes + 255) & ~(size_t)255;
        return p;
    };
    const size_t DD = (size_t)Dc * Dc, DFF = (size_t)Dc * FFc;
    unsigned short* wqT = (unsigned short*)carve(Lc * DD * 2);
    unsigned short* wkT = (unsigned short*)carve(Lc * DD * 2);
    unsigned short* wvT = (unsigned short*)carve(Lc * DD * 2);
    unsigned short* woT = (unsigned short*)carve(Lc * DD * 2);
    unsigned short* w1T = (unsigned short*)carve(Lc * DFF * 2);
    unsigned short* w2T = (unsigned short*)carve(Lc * DFF * 2);
    const size_t BSD = (size_t)Bc * Sc * Dc;
    float*          x      = (float*)         carve(BSD * 4);
    unsigned short* yh     = (unsigned short*)carve(BSD * 2);
    unsigned short* qh     = (unsigned short*)carve(BSD * 2);
    unsigned short* kh     = (unsigned short*)carve(BSD * 2);
    unsigned short* vh     = (unsigned short*)carve(BSD * 2);
    unsigned short* vt     = (unsigned short*)carve(BSD * 2);
    float*          p      = (float*)         carve((size_t)Bc * Hc * Sc * Rc * 4);
    float*          scores = (float*)         carve((size_t)Bc * Hc * Sc * Sc * 4);
    unsigned short* attnh  = (unsigned short*)carve((size_t)Bc * Hc * Sc * Sc * 2);
    float*          wbuf   = (float*)         carve((size_t)Bc * Hc * Sc * Rc * 4);
    float*          obuf   = (float*)         carve(BSD * 4);
    unsigned short* oh     = (unsigned short*)carve(BSD * 2);
    unsigned short* ff1    = (unsigned short*)carve((size_t)Bc * Sc * FFc * 2);

    dim3 gblk(32, 4);

    // weight conversion (every call; no cached state)
    for (int l = 0; l < Lc; ++l) {
        tconv_kernel<<<(int)((DD + 255) / 256), 256, 0, stream>>>(wqT + l * DD, Wq + l * DD, Dc, Dc);
        tconv_kernel<<<(int)((DD + 255) / 256), 256, 0, stream>>>(wkT + l * DD, Wk + l * DD, Dc, Dc);
        tconv_kernel<<<(int)((DD + 255) / 256), 256, 0, stream>>>(wvT + l * DD, Wv + l * DD, Dc, Dc);
        tconv_kernel<<<(int)((DD + 255) / 256), 256, 0, stream>>>(woT + l * DD, Wo + l * DD, Dc, Dc);
        tconv_kernel<<<(int)((DFF + 255) / 256), 256, 0, stream>>>(w1T + l * DFF, W1 + l * DFF, Dc, FFc);
        tconv_kernel<<<(int)((DFF + 255) / 256), 256, 0, stream>>>(w2T + l * DFF, W2 + l * DFF, FFc, Dc);
    }

    copy_f32_kernel<<<(int)((BSD + 255) / 256), 256, 0, stream>>>(x, enc, (int)BSD);

    const int M = Bc * Sc;   // 1024
    for (int l = 0; l < Lc; ++l) {
        // pre-norm
        layernorm_kernel<<<M, 256, 0, stream>>>(x, ln1_g + l * Dc, ln1_b + l * Dc, yh, nullptr);
        // QKV projections (bf16 WMMA, double-buffered async-LDS staging)
        gemm_wmma_kernel<<<dim3(M / 64, Dc / 64), gblk, 0, stream>>>(
            yh, Dc, wqT + l * DD, Dc, bq + l * Dc, nullptr, 0, 0, qh, Dc, M, Dc, Dc, 0);
        gemm_wmma_kernel<<<dim3(M / 64, Dc / 64), gblk, 0, stream>>>(
            yh, Dc, wkT + l * DD, Dc, bk + l * Dc, nullptr, 0, 0, kh, Dc, M, Dc, Dc, 0);
        gemm_wmma_kernel<<<dim3(M / 64, Dc / 64), gblk, 0, stream>>>(
            yh, Dc, wvT + l * DD, Dc, bv + l * Dc, nullptr, 0, 0, vh, Dc, M, Dc, Dc, 0);
        vt_kernel<<<(int)((BSD + 255) / 256), 256, 0, stream>>>(vt, vh);
        // relational key table p = q @ rel_k_emb^T
        pq_kernel<<<Bc * Hc * Sc, 64, 0, stream>>>(qh, relk + (size_t)l * Rc * DKc, p);
        // scores (WMMA QK^T fused with p-gather, mask, scale)
        scores_wmma_kernel<<<dim3(Sc / 16, Sc / 256, Bc * Hc), gblk, 0, stream>>>(
            qh, kh, p, relations, mask, scores);
        softmax_kernel<<<Bc * Hc * Sc, 256, 0, stream>>>(scores, attnh);
        bucket_kernel<<<Bc * Hc * Sc, 128, 0, stream>>>(scores, relations, wbuf);
        // attn @ V (WMMA)
        av_wmma_kernel<<<dim3(Sc / 64, 1, Bc * Hc), gblk, 0, stream>>>(attnh, vt, obuf);
        // + w @ rel_v_emb, scatter heads back, convert to bf16
        orv_kernel<<<(int)((BSD + 255) / 256), 256, 0, stream>>>(
            obuf, wbuf, relv + (size_t)l * Rc * DKc, oh);
        // O-projection with residual add into x
        gemm_wmma_kernel<<<dim3(M / 64, Dc / 64), gblk, 0, stream>>>(
            oh, Dc, woT + l * DD, Dc, bo + l * Dc, x, Dc, 1, nullptr, 0, M, Dc, Dc, 0);
        // FFN
        layernorm_kernel<<<M, 256, 0, stream>>>(x, ln2_g + l * Dc, ln2_b + l * Dc, yh, nullptr);
        gemm_wmma_kernel<<<dim3(M / 64, FFc / 64), gblk, 0, stream>>>(
            yh, Dc, w1T + l * DFF, Dc, b1 + l * FFc, nullptr, 0, 0, ff1, FFc, M, FFc, Dc, 1);
        gemm_wmma_kernel<<<dim3(M / 64, Dc / 64), gblk, 0, stream>>>(
            ff1, FFc, w2T + l * DFF, FFc, b2 + l * Dc, x, Dc, 1, nullptr, 0, M, Dc, FFc, 0);
    }

    // final LayerNorm -> fp32 output
    layernorm_kernel<<<M, 256, 0, stream>>>(x, lnf_g, lnf_b, nullptr, (float*)d_out);
    (void)in_sizes; (void)n_in; (void)out_size; (void)ws_size;
}
